// FlexAttentionEncoder_61186104099480
// MI455X (gfx1250) — matrix operand order repaired and checked by reading, not compiled, not run on hardware
//
#include <hip/hip_runtime.h>
#include <hip/hip_bf16.h>

#define LAYERS 2
#define DMODEL 768
#define NHEAD  12
#define HDIM   64
#define BATCH  2
#define SEQ    2048
#define FFD    3072
#define WINSZ  512
#define ROWS   (BATCH * SEQ)   // 4096

typedef __attribute__((ext_vector_type(8)))  _Float16 v8h;
typedef __attribute__((ext_vector_type(16))) _Float16 v16h;
typedef __attribute__((ext_vector_type(8)))  float    v8f;

// ---- A-matrix fragment (16x32 f16): lane<16 -> row=lane&15, K={koff..koff+7, koff+16..koff+23}
// with koff=(lane>>4)*8.  Two 16-byte loads.
__device__ __forceinline__ v16h load_a_frag(const _Float16* p) {
  v16h t;
  *(v8h*)&t       = *(const v8h*)p;
  *((v8h*)&t + 1) = *(const v8h*)(p + 16);
  return t;
}

// =====================  GEMM: C[M,N] = A[M,K] * W[N,K]^T + bias  =====================
// Block = 256 threads = 8 waves (2x4), wave tile 32x32 (2x2 WMMA), block tile 64x128.
// TRANSV: scatter f16 output into per-head-transposed layout [B][NHEAD][HDIM][SEQ].
template <bool RELU, bool OUTF16, bool TRANSV>
__global__ __launch_bounds__(256) void gemm_tn(
    const _Float16* __restrict__ A, const _Float16* __restrict__ Bw,
    const float* __restrict__ bias, float* __restrict__ Cf, _Float16* __restrict__ Ch,
    int M, int N, int K)
{
  const int wid  = threadIdx.x >> 5;
  const int lane = threadIdx.x & 31;
  const int wm = wid >> 2, wn = wid & 3;
  const int m0 = blockIdx.x * 64  + wm * 32;
  const int n0 = blockIdx.y * 128 + wn * 32;
  const int r    = lane & 15;
  const int koff = (lane >> 4) << 3;   // A interleave
  const int kblk = (lane >> 4) << 4;   // contiguous half (unused; B loads use koff)

  v8f acc[2][2] = {};
  for (int k0 = 0; k0 < K; k0 += 32) {
    v16h a[2], b[2];
#pragma unroll
    for (int i = 0; i < 2; ++i) {
      a[i] = load_a_frag(A + (size_t)(m0 + i * 16 + r) * K + k0 + koff);
      b[i] = load_a_frag(Bw + (size_t)(n0 + i * 16 + r) * K + k0 + koff);
    }
#pragma unroll
    for (int i = 0; i < 2; ++i)
#pragma unroll
      for (int j = 0; j < 2; ++j)
        acc[i][j] = __builtin_amdgcn_wmma_f32_16x16x32_f16(
            false, a[i], false, b[j], (short)0, acc[i][j], false, false);
  }
  // epilogue: C layout -> row = t + 8*(lane>>4), col = lane&15
  const int mrb = (lane >> 4) << 3;
#pragma unroll
  for (int i = 0; i < 2; ++i)
#pragma unroll
    for (int j = 0; j < 2; ++j) {
      const int nn = n0 + j * 16 + r;
      const float bv = bias ? bias[nn] : 0.0f;
#pragma unroll
      for (int t = 0; t < 8; ++t) {
        const int mm = m0 + i * 16 + mrb + t;
        float v = acc[i][j][t] + bv;
        if (RELU) v = v > 0.0f ? v : 0.0f;
        if (TRANSV) {
          // token mm = (b, s), channel nn = (h, d) -> Vt[b][h][d][s]
          const int bb = mm >> 11, ss = mm & (SEQ - 1);
          const int hh = nn >> 6,  dd = nn & (HDIM - 1);
          Ch[(((size_t)bb * NHEAD + hh) * HDIM + dd) * SEQ + ss] = (_Float16)v;
        } else if (OUTF16) {
          Ch[(size_t)mm * N + nn] = (_Float16)v;
        } else {
          Cf[(size_t)mm * N + nn] = v;
        }
      }
    }
}

// =====================  Sliding-window causal attention  =====================
// One wave per (batch, head, 16-query tile).  Two-pass softmax.
// Scores computed TRANSPOSED (St = K_tile x Q^T) so the St accumulator layout
// matches the A-fragment layout of P for the P x V WMMA with no shuffles.
// V is pre-transposed per head: Vt[b][h][d][s] -> vector B-fragment loads.
__global__ __launch_bounds__(256) void attn_kernel(
    const _Float16* __restrict__ Q, const _Float16* __restrict__ Kt,
    const _Float16* __restrict__ Vt, _Float16* __restrict__ O)
{
  const int widv = blockIdx.x * (blockDim.x >> 5) + (threadIdx.x >> 5);
  const int wid  = __builtin_amdgcn_readfirstlane(widv);   // wave-uniform -> SGPR
  const int qt = wid % (SEQ / 16);
  const int bh = wid / (SEQ / 16);
  const int h = bh % NHEAD, b = bh / NHEAD;
  const int q0 = qt * 16;
  const int lane = threadIdx.x & 31;
  const int r    = lane & 15;
  const int koff = (lane >> 4) << 3;
  const int kblk = (lane >> 4) << 4;
  const int col0 = h * HDIM;
  const size_t baseRow = (size_t)b * SEQ;
  const _Float16* __restrict__ Vhead = Vt + ((size_t)bh * HDIM) * SEQ;
  const float scale = 0.125f;  // 1/sqrt(64)

  // Q as B-operand fragments (two 32-wide steps over HDIM=64)
  v16h qf[2];
#pragma unroll
  for (int s2 = 0; s2 < 2; ++s2)
    qf[s2] = load_a_frag(Q + (baseRow + q0 + r) * DMODEL + col0 + s2 * 32 + koff);

  int kstart = q0 - WINSZ; if (kstart < 0) kstart = 0;
  kstart = __builtin_amdgcn_readfirstlane(kstart);

  // ---- pass 1: row max (per lane: query = lane&15)
  float mx = -1e30f;
  for (int kb = kstart; kb <= q0; kb += 16) {
    v8f st = {};
#pragma unroll
    for (int s2 = 0; s2 < 2; ++s2) {
      v16h kf = load_a_frag(Kt + (baseRow + kb + r) * DMODEL + col0 + s2 * 32 + koff);
      st = __builtin_amdgcn_wmma_f32_16x16x32_f16(false, kf, false, qf[s2],
                                                  (short)0, st, false, false);
    }
#pragma unroll
    for (int t = 0; t < 8; ++t) {
      const int k = kb + t + koff;     // key row in St
      const int q = q0 + r;            // query column in St
      const float sv = (q >= k && q - k <= WINSZ) ? st[t] * scale : -1e30f;
      mx = fmaxf(mx, sv);
    }
  }
  mx = fmaxf(mx, __shfl_xor(mx, 16, 32));

  // ---- pass 2: P = exp(s - mx), O += P x V, l accumulation
  float lsum = 0.0f;
  v8f o[4] = {};
  for (int kb = kstart; kb <= q0; kb += 32) {
    v16h pf;
#pragma unroll
    for (int half = 0; half < 2; ++half) {
      const int kbb = kb + half * 16;
      if (kbb <= q0) {               // scalar (SGPR) condition
        v8f st = {};
#pragma unroll
        for (int s2 = 0; s2 < 2; ++s2) {
          v16h kf = load_a_frag(Kt + (baseRow + kbb + r) * DMODEL + col0 + s2 * 32 + koff);
          st = __builtin_amdgcn_wmma_f32_16x16x32_f16(false, kf, false, qf[s2],
                                                      (short)0, st, false, false);
        }
#pragma unroll
        for (int t = 0; t < 8; ++t) {
          const int k = kbb + t + koff;
          const int q = q0 + r;
          const float sv = (q >= k && q - k <= WINSZ) ? st[t] * scale : -1e30f;
          const float p = __expf(sv - mx);
          lsum += p;
          pf[half * 8 + t] = (_Float16)p;
        }
      } else {
#pragma unroll
        for (int t = 0; t < 8; ++t) pf[half * 8 + t] = (_Float16)0.0f;
      }
    }
    // V as B-operand: lane holds column d, key rows {koff..koff+7, koff+16..koff+23} -> two 16B loads
#pragma unroll
    for (int j = 0; j < 4; ++j) {
      const v16h vf = load_a_frag(Vhead + (size_t)(j * 16 + r) * SEQ + kb + koff);
      o[j] = __builtin_amdgcn_wmma_f32_16x16x32_f16(false, pf, false, vf,
                                                    (short)0, o[j], false, false);
    }
  }
  lsum += __shfl_xor(lsum, 16, 32);
  const float linv = 1.0f / lsum;

#pragma unroll
  for (int j = 0; j < 4; ++j)
#pragma unroll
    for (int t = 0; t < 8; ++t) {
      const int qrow = t + koff;                   // O row = query
      const float sc = __shfl(linv, qrow, 32);     // lane qrow holds l[qrow]
      O[(baseRow + q0 + qrow) * DMODEL + col0 + j * 16 + r] =
          (_Float16)(o[j][t] * sc);
    }
}

// =====================  residual + LayerNorm (+ f16 recast)  =====================
__global__ __launch_bounds__(256) void add_ln_kernel(
    const float* __restrict__ X, const float* __restrict__ T,
    const float* __restrict__ g, const float* __restrict__ be,
    float* __restrict__ Xout, _Float16* __restrict__ Hout)
{
  const int row  = blockIdx.x * (blockDim.x >> 5) + (threadIdx.x >> 5);
  const int lane = threadIdx.x & 31;
  const float* px = X + (size_t)row * DMODEL;
  const float* pt = T + (size_t)row * DMODEL;
  float vals[DMODEL / 32];
  float s = 0.0f, ss = 0.0f;
#pragma unroll
  for (int i = 0; i < DMODEL / 32; ++i) {
    const float v = px[lane + i * 32] + pt[lane + i * 32];
    vals[i] = v; s += v; ss += v * v;
  }
#pragma unroll
  for (int off = 16; off; off >>= 1) {
    s  += __shfl_xor(s,  off, 32);
    ss += __shfl_xor(ss, off, 32);
  }
  const float mu  = s / (float)DMODEL;
  const float var = ss / (float)DMODEL - mu * mu;
  const float rstd = rsqrtf(var + 1e-5f);
#pragma unroll
  for (int i = 0; i < DMODEL / 32; ++i) {
    const int c = lane + i * 32;
    const float y = (vals[i] - mu) * rstd * g[c] + be[c];
    Xout[(size_t)row * DMODEL + c] = y;
    Hout[(size_t)row * DMODEL + c] = (_Float16)y;
  }
}

__global__ void f32_to_f16_kernel(const float* __restrict__ in,
                                  _Float16* __restrict__ out, size_t n)
{
  size_t i = (size_t)blockIdx.x * blockDim.x + threadIdx.x;
  const size_t stride = (size_t)gridDim.x * blockDim.x;
  for (; i < n; i += stride) out[i] = (_Float16)in[i];
}

// ===========================================================================
extern "C" void kernel_launch(void* const* d_in, const int* in_sizes, int n_in,
                              void* d_out, int out_size, void* d_ws, size_t ws_size,
                              hipStream_t stream)
{
  (void)in_sizes; (void)n_in; (void)out_size; (void)ws_size;
  const float* x   = (const float*)d_in[0];
  const float* Wq  = (const float*)d_in[1];
  const float* bq  = (const float*)d_in[2];
  const float* Wk  = (const float*)d_in[3];
  const float* bk  = (const float*)d_in[4];
  const float* Wv  = (const float*)d_in[5];
  const float* bv  = (const float*)d_in[6];
  const float* Wo  = (const float*)d_in[7];
  const float* bo  = (const float*)d_in[8];
  const float* W1  = (const float*)d_in[9];
  const float* b1  = (const float*)d_in[10];
  const float* W2  = (const float*)d_in[11];
  const float* b2  = (const float*)d_in[12];
  const float* gm1 = (const float*)d_in[13];
  const float* bt1 = (const float*)d_in[14];
  const float* gm2 = (const float*)d_in[15];
  const float* bt2 = (const float*)d_in[16];

  char* ws = (char*)d_ws;
  size_t off = 0;
  auto alloc = [&](size_t bytes) { void* p = ws + off; off += (bytes + 255) & ~(size_t)255; return p; };

  float*    X   = (float*)   alloc((size_t)ROWS * DMODEL * 4);
  float*    T   = (float*)   alloc((size_t)ROWS * DMODEL * 4);
  _Float16* Xh  = (_Float16*)alloc((size_t)ROWS * DMODEL * 2);
  _Float16* Qh  = (_Float16*)alloc((size_t)ROWS * DMODEL * 2);
  _Float16* Kh  = (_Float16*)alloc((size_t)ROWS * DMODEL * 2);
  _Float16* Vth = (_Float16*)alloc((size_t)ROWS * DMODEL * 2 + 256);  // pad: masked-tail overread
  _Float16* Ah  = (_Float16*)alloc((size_t)ROWS * DMODEL * 2);
  _Float16* Hff = (_Float16*)alloc((size_t)ROWS * FFD * 2);
  _Float16* WqH = (_Float16*)alloc((size_t)LAYERS * DMODEL * DMODEL * 2);
  _Float16* WkH = (_Float16*)alloc((size_t)LAYERS * DMODEL * DMODEL * 2);
  _Float16* WvH = (_Float16*)alloc((size_t)LAYERS * DMODEL * DMODEL * 2);
  _Float16* WoH = (_Float16*)alloc((size_t)LAYERS * DMODEL * DMODEL * 2);
  _Float16* W1H = (_Float16*)alloc((size_t)LAYERS * FFD * DMODEL * 2);
  _Float16* W2H = (_Float16*)alloc((size_t)LAYERS * DMODEL * FFD * 2);

  const size_t nDD = (size_t)LAYERS * DMODEL * DMODEL;
  const size_t nFD = (size_t)LAYERS * FFD * DMODEL;
  f32_to_f16_kernel<<<1024, 256, 0, stream>>>(Wq, WqH, nDD);
  f32_to_f16_kernel<<<1024, 256, 0, stream>>>(Wk, WkH, nDD);
  f32_to_f16_kernel<<<1024, 256, 0, stream>>>(Wv, WvH, nDD);
  f32_to_f16_kernel<<<1024, 256, 0, stream>>>(Wo, WoH, nDD);
  f32_to_f16_kernel<<<1024, 256, 0, stream>>>(W1, W1H, nFD);
  f32_to_f16_kernel<<<1024, 256, 0, stream>>>(W2, W2H, nFD);

  hipMemcpyAsync(X, x, (size_t)ROWS * DMODEL * 4, hipMemcpyDeviceToDevice, stream);
  f32_to_f16_kernel<<<1024, 256, 0, stream>>>(x, Xh, (size_t)ROWS * DMODEL);

  const dim3 gD(ROWS / 64, DMODEL / 128);   // (64, 6)
  const dim3 gF(ROWS / 64, FFD / 128);      // (64, 24)
  const int attnBlocks = (BATCH * NHEAD * (SEQ / 16)) / 8;  // 384
  const int lnBlocks   = ROWS / 8;                          // 512

  for (int l = 0; l < LAYERS; ++l) {
    const size_t oDD = (size_t)l * DMODEL * DMODEL;
    const size_t oFD = (size_t)l * FFD * DMODEL;
    // QKV projections (f16 out; V written head-transposed)
    gemm_tn<false, true, false><<<gD, 256, 0, stream>>>(Xh, WqH + oDD, bq + l * DMODEL,
                                                        nullptr, Qh, ROWS, DMODEL, DMODEL);
    gemm_tn<false, true, false><<<gD, 256, 0, stream>>>(Xh, WkH + oDD, bk + l * DMODEL,
                                                        nullptr, Kh, ROWS, DMODEL, DMODEL);
    gemm_tn<false, true, true ><<<gD, 256, 0, stream>>>(Xh, WvH + oDD, bv + l * DMODEL,
                                                        nullptr, Vth, ROWS, DMODEL, DMODEL);
    // sliding-window attention
    attn_kernel<<<attnBlocks, 256, 0, stream>>>(Qh, Kh, Vth, Ah);
    // output projection (f32 out) + residual/LN
    gemm_tn<false, false, false><<<gD, 256, 0, stream>>>(Ah, WoH + oDD, bo + l * DMODEL,
                                                         T, nullptr, ROWS, DMODEL, DMODEL);
    add_ln_kernel<<<lnBlocks, 256, 0, stream>>>(X, T, gm1 + l * DMODEL, bt1 + l * DMODEL, X, Xh);
    // FFN
    gemm_tn<true, true, false><<<gF, 256, 0, stream>>>(Xh, W1H + oFD, b1 + l * FFD,
                                                       nullptr, Hff, ROWS, FFD, DMODEL);
    gemm_tn<false, false, false><<<gD, 256, 0, stream>>>(Hff, W2H + oFD, b2 + l * DMODEL,
                                                         T, nullptr, ROWS, DMODEL, FFD);
    add_ln_kernel<<<lnBlocks, 256, 0, stream>>>(X, T, gm2 + l * DMODEL, bt2 + l * DMODEL, X, Xh);
  }

  hipMemcpyAsync(d_out, X, (size_t)ROWS * DMODEL * 4, hipMemcpyDeviceToDevice, stream);
}
